// PolNetLSTM_87969520157378
// MI455X (gfx1250) — matrix operand order repaired and checked by reading, not compiled, not run on hardware
//
#include <hip/hip_runtime.h>
#include <hip/hip_bf16.h>
#include <stdint.h>

// ---------------------------------------------------------------------------
// Problem constants (from reference): T=512, B=64, OB=128, AC=32, H=1024, C=512
// ---------------------------------------------------------------------------
#define TT   512
#define BB   64
#define OBS  128
#define ACD  32
#define HH   1024
#define CC   512
#define G4   2048   // 4*C

typedef __attribute__((ext_vector_type(16))) __bf16    v16bf;
typedef __attribute__((ext_vector_type(8)))  float     v8f;
typedef __attribute__((ext_vector_type(4)))  uint32_t  v4u;
typedef __attribute__((ext_vector_type(8)))  uint32_t  v8u;

union FragBF {
    v16bf v;
    uint4 q[2];
};

__device__ __forceinline__ unsigned short f2bf(float f) {
    union { float f; uint32_t u; } a; a.f = f;
    uint32_t u = a.u;
    u += 0x7FFFu + ((u >> 16) & 1u);   // round-to-nearest-even
    return (unsigned short)(u >> 16);
}
__device__ __forceinline__ float bf2f(unsigned short b) {
    union { float f; uint32_t u; } a; a.u = ((uint32_t)b) << 16;
    return a.f;
}
__device__ __forceinline__ float sigmoidf_(float x) {
    return 1.0f / (1.0f + __expf(-x));
}
__device__ __forceinline__ v8f wmma_bf16(const FragBF& A, const FragBF& B, v8f C) {
    return __builtin_amdgcn_wmma_f32_16x16x32_bf16(false, A.v, false, B.v,
                                                   (short)0, C, false, false);
}

// ---------------------------------------------------------------------------
// Tensor Data Mover (TDM) helpers: build D# descriptor groups (uniform values
// -> SGPRs) and issue tensor_load_to_lds via inline asm (portable across
// toolchains per bridge doc; 2-operand form = tensors up to 2D).
// ---------------------------------------------------------------------------
__device__ __forceinline__ v4u tdm_g0(uint32_t lds_byte_off, const void* gaddr) {
    uint64_t ga = (uint64_t)gaddr;
    v4u g;
    g[0] = 1u;                                              // count=1, no gather
    g[1] = lds_byte_off;                                    // lds_addr
    g[2] = (uint32_t)ga;                                    // global_addr[31:0]
    g[3] = (uint32_t)((ga >> 32) & 0x1FFFFFFu) | (2u << 30);// addr[56:32] | type=2
    return g;
}
// 2D tile: data_size_log2: 0=1B,1=2B,2=4B; dims/tiles in elements.
__device__ __forceinline__ v8u tdm_g1_2d(uint32_t ds_log2, uint32_t dim0, uint32_t dim1,
                                         uint32_t tile0, uint32_t tile1, uint64_t stride0) {
    v8u g;
    g[0] = ds_log2 << 16;                                   // wg_mask=0, data_size
    g[1] = (dim0 & 0xFFFFu) << 16;                          // tensor_dim0[15:0]
    g[2] = ((dim0 >> 16) & 0xFFFFu) | ((dim1 & 0xFFFFu) << 16);
    g[3] = ((dim1 >> 16) & 0xFFFFu) | ((tile0 & 0xFFFFu) << 16);  // tile_dim0
    g[4] = (tile1 & 0xFFFFu);                               // tile_dim1 (0 => 1D)
    g[5] = (uint32_t)stride0;                               // tensor_dim0_stride lo
    g[6] = (uint32_t)((stride0 >> 32) & 0xFFFFu);           // stride0 hi, stride1 lo=0
    g[7] = 0u;
    return g;
}
__device__ __forceinline__ void tdm_load_to_lds(v4u g0, v8u g1) {
    asm volatile("tensor_load_to_lds %0, %1" :: "s"(g0), "s"(g1) : "memory");
}
__device__ __forceinline__ uint32_t lds_offset_of(const void* p) {
    return (uint32_t)(uintptr_t)p;   // low 32 bits of generic addr = LDS byte offset
}

// ---------------------------------------------------------------------------
// Kernel 1: fp32 -> bf16 conversion
// ---------------------------------------------------------------------------
__global__ void k_cvt_bf16(const float* __restrict__ src,
                           unsigned short* __restrict__ dst, int n) {
    int i = blockIdx.x * blockDim.x + threadIdx.x;
    if (i < n) dst[i] = f2bf(src[i]);
}

// ---------------------------------------------------------------------------
// Kernel 2: generic bf16 WMMA GEMM  Y[M,N] = act(X[M,K] * W[N,K]^T + bias[N])
// Block = 128 threads (4 waves); wave computes 16(M) x 64(N).
// All four B fragments are loaded before the WMMA burst so the loads clause
// and drain with partial s_wait_loadcnt while the matrix pipe runs.
// ---------------------------------------------------------------------------
__global__ __launch_bounds__(128) void k_gemm_bf16(
    const unsigned short* __restrict__ X,
    const unsigned short* __restrict__ W,
    const float* __restrict__ bias,
    int K, int Nld,
    float* __restrict__ Yf,
    unsigned short* __restrict__ Ybf)
{
    const int lane = threadIdx.x & 31;
    const int wave = threadIdx.x >> 5;
    const int r    = lane & 15;
    const int sel  = lane >> 4;
    const int m0   = blockIdx.y * 64 + wave * 16;
    const int n0   = blockIdx.x * 64;

    v8f acc[4] = {v8f{}, v8f{}, v8f{}, v8f{}};

    const int ksteps = K >> 5;
    for (int ks = 0; ks < ksteps; ++ks) {
        FragBF a;
        const int kb = ks * 32 + sel * 8;
        const unsigned short* xp = X + (size_t)(m0 + r) * K + kb;
        a.q[0] = *(const uint4*)(xp);
        a.q[1] = *(const uint4*)(xp + 16);

        const int kbb = ks * 32 + sel * 8;
        FragBF b[4];
        #pragma unroll
        for (int nt = 0; nt < 4; ++nt) {
            const unsigned short* wp = W + (size_t)(n0 + nt * 16 + r) * K + kbb;
            __builtin_prefetch(wp + 64, 0, 1);   // global_prefetch_b8
            b[nt].q[0] = *(const uint4*)(wp);
            b[nt].q[1] = *(const uint4*)(wp + 16);
        }
        #pragma unroll
        for (int nt = 0; nt < 4; ++nt)
            acc[nt] = wmma_bf16(a, b[nt], acc[nt]);
    }

    #pragma unroll
    for (int nt = 0; nt < 4; ++nt) {
        const int n  = n0 + nt * 16 + r;
        const float bv = bias[n];
        #pragma unroll
        for (int rr = 0; rr < 8; ++rr) {
            const int row = m0 + sel * 8 + rr;
            float v = acc[nt][rr] + bv;
            if (Ybf) {
                v = v > 0.0f ? v : 0.0f;                 // relu
                Ybf[(size_t)row * Nld + n] = f2bf(v);
            } else {
                Yf[(size_t)row * Nld + n] = v;
            }
        }
    }
}

// ---------------------------------------------------------------------------
// Kernel 3: persistent LSTM scan. ONE workgroup, 1024 threads = 32 waves.
// h: LDS bf16 (64x512 = 64KB). c: registers. W_mean staged to LDS by the TDM
// once; per-step mask vector double-buffered in LDS via TDM, synchronized
// with s_wait_tensorcnt.
// ---------------------------------------------------------------------------
__global__ __launch_bounds__(1024, 1) void k_lstm_scan(
    const float* __restrict__ gx,            // [T*B, 4C] fp32 (precomputed)
    const float* __restrict__ masks,         // [T*B]
    const float* __restrict__ h0,            // [B,C]
    const float* __restrict__ c0,            // [B,C]
    const unsigned short* __restrict__ Whh,  // [2048,512] bf16
    const unsigned short* __restrict__ Wmean,// [32,512]   bf16 (TDM source)
    const float* __restrict__ b_hh,          // [2048]
    const float* __restrict__ b_mean,        // [32]
    float* __restrict__ means_out,           // [T*B*AC]
    float* __restrict__ hT_out,              // [B*C]
    float* __restrict__ cT_out)              // [B*C]
{
    __shared__ unsigned short h_lds[BB * CC];        // 64KB
    __shared__ unsigned short wmean_lds[ACD * CC];   // 32KB, filled by TDM
    __shared__ float mask_lds[2][BB];                // double-buffered via TDM

    const int tid  = threadIdx.x;
    const int lane = tid & 31;
    const int wave = tid >> 5;     // 0..31
    const int r    = lane & 15;
    const int sel  = lane >> 4;
    const int j0   = wave * 16;    // column group (within C) owned by this wave

    // ---- TDM: stage W_mean (32x512 bf16) and masks[0]; wave 0 only ----
    if (wave == 0) {
        tdm_load_to_lds(tdm_g0(lds_offset_of(wmean_lds), Wmean),
                        tdm_g1_2d(/*2B*/1, CC, ACD, CC, ACD, CC));
        tdm_load_to_lds(tdm_g0(lds_offset_of(&mask_lds[0][0]), masks),
                        tdm_g1_2d(/*4B*/2, BB, 1, BB, 0, BB));
    }

    // ---- init h (LDS) and c (regs) from h0/c0 ----
    for (int i = tid; i < BB * CC; i += 1024) h_lds[i] = f2bf(h0[i]);

    float creg[4][8];   // c[(mt*16 + sel*8 + rr), j0 + r]
    float hreg[4][8];   // staging for h_new before LDS writeback
    #pragma unroll
    for (int mt = 0; mt < 4; ++mt)
        #pragma unroll
        for (int rr = 0; rr < 8; ++rr)
            creg[mt][rr] = c0[(size_t)(mt * 16 + sel * 8 + rr) * CC + j0 + r];

    // ---- biases ----
    float bhh[4];
    #pragma unroll
    for (int g = 0; g < 4; ++g) bhh[g] = b_hh[g * CC + j0 + r];
    const float bmean = (wave < 8) ? b_mean[(wave & 1) * 16 + r] : 0.0f;

    __syncthreads();

    for (int t = 0; t < TT; ++t) {
        // ---- wait for TDM mask[t] (and W_mean on t==0), prefetch mask[t+1] ----
        if (wave == 0) {
            __builtin_amdgcn_s_wait_tensorcnt(0);
            if (t + 1 < TT)
                tdm_load_to_lds(tdm_g0(lds_offset_of(&mask_lds[(t + 1) & 1][0]),
                                       masks + (size_t)(t + 1) * BB),
                                tdm_g1_2d(2, BB, 1, BB, 0, BB));
        }
        __syncthreads();
        const float* mrow = mask_lds[t & 1];

        // ---- masks: h *= (1-m), c *= (1-m) ----
        for (int i = tid; i < BB * CC; i += 1024) {
            float hv = bf2f(h_lds[i]);
            h_lds[i] = f2bf(hv * (1.0f - mrow[i >> 9]));   // row = i / 512
        }
        __syncthreads();
        #pragma unroll
        for (int mt = 0; mt < 4; ++mt)
            #pragma unroll
            for (int rr = 0; rr < 8; ++rr)
                creg[mt][rr] *= (1.0f - mrow[mt * 16 + sel * 8 + rr]);

        // ---- gate GEMM + cell update, 4 M-tiles of 16 rows ----
        #pragma unroll
        for (int mt = 0; mt < 4; ++mt) {
            const int m0 = mt * 16;
            v8f acc[4];
            // seed accumulators with gx[t] + b_hh
            #pragma unroll
            for (int g = 0; g < 4; ++g) {
                const float* gxp = gx + (size_t)(t * BB) * G4 + g * CC + j0 + r;
                v8f a;
                #pragma unroll
                for (int rr = 0; rr < 8; ++rr)
                    a[rr] = gxp[(size_t)(m0 + sel * 8 + rr) * G4] + bhh[g];
                acc[g] = a;
            }
            // K loop over C=512 (16 steps of 32): batch all loads, then WMMA burst
            for (int ks = 0; ks < 16; ++ks) {
                FragBF af;
                const int kb = ks * 32 + sel * 8;
                af.q[0] = *(const uint4*)(&h_lds[(m0 + r) * CC + kb]);
                af.q[1] = *(const uint4*)(&h_lds[(m0 + r) * CC + kb + 16]);
                const int kbb = ks * 32 + sel * 8;
                FragBF bf_[4];
                #pragma unroll
                for (int g = 0; g < 4; ++g) {
                    const unsigned short* wp = Whh + (size_t)(g * CC + j0 + r) * CC + kbb;
                    __builtin_prefetch(wp + 32, 0, 1);
                    bf_[g].q[0] = *(const uint4*)(wp);
                    bf_[g].q[1] = *(const uint4*)(wp + 16);
                }
                #pragma unroll
                for (int g = 0; g < 4; ++g)
                    acc[g] = wmma_bf16(af, bf_[g], acc[g]);
            }
            // elementwise LSTM cell (torch gate order i,f,g,o)
            #pragma unroll
            for (int rr = 0; rr < 8; ++rr) {
                const float iv = sigmoidf_(acc[0][rr]);
                const float fv = sigmoidf_(acc[1][rr]);
                const float gv = tanhf(acc[2][rr]);
                const float ov = sigmoidf_(acc[3][rr]);
                const float cv = fv * creg[mt][rr] + iv * gv;
                creg[mt][rr] = cv;
                hreg[mt][rr] = ov * tanhf(cv);
            }
        }

        // ---- publish h_new to LDS (after all waves finished reading old h) ----
        __syncthreads();
        #pragma unroll
        for (int mt = 0; mt < 4; ++mt)
            #pragma unroll
            for (int rr = 0; rr < 8; ++rr)
                h_lds[(mt * 16 + sel * 8 + rr) * CC + j0 + r] = f2bf(hreg[mt][rr]);
        __syncthreads();

        // ---- mean head: [64,512] x [512,32], 8 tiles on waves 0..7.
        //      B fragments come from the TDM-staged W_mean in LDS. ----
        if (wave < 8) {
            const int mt = wave >> 1, nt = wave & 1;
            const int m0 = mt * 16, n0 = nt * 16;
            v8f acc = {};
            for (int ks = 0; ks < 16; ++ks) {
                FragBF af, bf_;
                const int kb = ks * 32 + sel * 8;
                af.q[0] = *(const uint4*)(&h_lds[(m0 + r) * CC + kb]);
                af.q[1] = *(const uint4*)(&h_lds[(m0 + r) * CC + kb + 16]);
                const int kbb = ks * 32 + sel * 8;
                const unsigned short* wp = &wmean_lds[(n0 + r) * CC + kbb];
                bf_.q[0] = *(const uint4*)(wp);
                bf_.q[1] = *(const uint4*)(wp + 16);
                acc = wmma_bf16(af, bf_, acc);
            }
            #pragma unroll
            for (int rr = 0; rr < 8; ++rr) {
                const int row = m0 + sel * 8 + rr;   // batch
                const int col = n0 + r;              // action dim
                means_out[((size_t)t * BB + row) * ACD + col] = tanhf(acc[rr] + bmean);
            }
        }
        __syncthreads();
    }

    // ---- final state writeback ----
    for (int i = tid; i < BB * CC; i += 1024) hT_out[i] = bf2f(h_lds[i]);
    #pragma unroll
    for (int mt = 0; mt < 4; ++mt)
        #pragma unroll
        for (int rr = 0; rr < 8; ++rr)
            cT_out[(size_t)(mt * 16 + sel * 8 + rr) * CC + j0 + r] = creg[mt][rr];
}

// ---------------------------------------------------------------------------
// Kernel 4: log_std broadcast
// ---------------------------------------------------------------------------
__global__ void k_logstd(const float* __restrict__ p, float* __restrict__ out, int n) {
    int i = blockIdx.x * blockDim.x + threadIdx.x;
    if (i < n) out[i] = p[i & (ACD - 1)];
}

// ---------------------------------------------------------------------------
// Host launcher
// ---------------------------------------------------------------------------
extern "C" void kernel_launch(void* const* d_in, const int* in_sizes, int n_in,
                              void* d_out, int out_size, void* d_ws, size_t ws_size,
                              hipStream_t stream) {
    (void)in_sizes; (void)n_in; (void)out_size; (void)ws_size;
    // inputs (setup_inputs order)
    const float* xs       = (const float*)d_in[0];   // [T,B,OB]
    const float* masks    = (const float*)d_in[1];   // [T,B,1]
    const float* h0       = (const float*)d_in[2];   // [B,C]
    const float* c0       = (const float*)d_in[3];   // [B,C]
    const float* W_in     = (const float*)d_in[4];   // [H,OB]
    const float* b_in     = (const float*)d_in[5];   // [H]
    const float* W_ih     = (const float*)d_in[6];   // [4C,H]
    const float* W_hh     = (const float*)d_in[7];   // [4C,C]
    const float* b_ih     = (const float*)d_in[8];   // [4C]
    const float* b_hh     = (const float*)d_in[9];   // [4C]
    const float* W_mean   = (const float*)d_in[10];  // [AC,C]
    const float* b_mean   = (const float*)d_in[11];  // [AC]
    const float* log_std  = (const float*)d_in[12];  // [AC]

    const size_t M = (size_t)TT * BB;                // 32768 rows

    // workspace layout (bytes)
    char* ws = (char*)d_ws;
    unsigned short* xs_bf = (unsigned short*)(ws);                       size_t off = M * OBS * 2;
    unsigned short* x_bf  = (unsigned short*)(ws + off);                 off += M * HH * 2;
    float*          gx    = (float*)(ws + off);                          off += M * G4 * 4;
    unsigned short* Win_b = (unsigned short*)(ws + off);                 off += (size_t)HH * OBS * 2;
    unsigned short* Wih_b = (unsigned short*)(ws + off);                 off += (size_t)G4 * HH * 2;
    unsigned short* Whh_b = (unsigned short*)(ws + off);                 off += (size_t)G4 * CC * 2;
    unsigned short* Wmn_b = (unsigned short*)(ws + off);                 off += (size_t)ACD * CC * 2;

    // output layout: means | log_std | hT | cT
    float* means_out  = (float*)d_out;
    float* logstd_out = means_out + M * ACD;
    float* hT_out     = logstd_out + M * ACD;
    float* cT_out     = hT_out + (size_t)BB * CC;

    // ---- 1. bf16 conversions ----
    auto cvt = [&](const float* s, unsigned short* d, size_t n) {
        k_cvt_bf16<<<(unsigned)((n + 255) / 256), 256, 0, stream>>>(s, d, (int)n);
    };
    cvt(xs,     xs_bf, M * OBS);
    cvt(W_in,   Win_b, (size_t)HH * OBS);
    cvt(W_ih,   Wih_b, (size_t)G4 * HH);
    cvt(W_hh,   Whh_b, (size_t)G4 * CC);
    cvt(W_mean, Wmn_b, (size_t)ACD * CC);

    // ---- 2. x = relu(xs @ W_in^T + b_in)  -> bf16 [M, H] ----
    {
        dim3 grid(HH / 64, (unsigned)(M / 64));
        k_gemm_bf16<<<grid, 128, 0, stream>>>(xs_bf, Win_b, b_in, OBS, HH,
                                              nullptr, x_bf);
    }

    // ---- 3. gx = x @ W_ih^T + b_ih  -> fp32 [M, 4C] ----
    {
        dim3 grid(G4 / 64, (unsigned)(M / 64));
        k_gemm_bf16<<<grid, 128, 0, stream>>>(x_bf, Wih_b, b_ih, HH, G4,
                                              gx, nullptr);
    }

    // ---- 4. sequential LSTM scan (single persistent workgroup) ----
    k_lstm_scan<<<1, 1024, 0, stream>>>(gx, masks, h0, c0, Whh_b, Wmn_b,
                                        b_hh, b_mean, means_out, hT_out, cT_out);

    // ---- 5. log_std broadcast ----
    {
        const size_t n = M * ACD;
        k_logstd<<<(unsigned)((n + 255) / 256), 256, 0, stream>>>(log_std, logstd_out, (int)n);
    }
}
